// ALSTMCell_64836826300984
// MI455X (gfx1250) — matrix operand order repaired and checked by reading, not compiled, not run on hardware
//
#include <hip/hip_runtime.h>
#include <hip/hip_bf16.h>
#include <math.h>

// ---------------------------------------------------------------------------
// ALSTM (attention-LSTM) scan for MI455X / gfx1250, wave32 + WMMA bf16.
// B=256, T=512, D=256, U=512.
//
//  * 16 blocks x 512 threads (16 wave32 waves); block owns a 16-row batch
//    tile (M=16 == WMMA M) for the entire T loop; h/c state lives in LDS.
//  * All matmuls via v_wmma_f32_16x16x32_bf16 (fp32 accumulate).
//  * Weights pre-packed fp32->bf16 into d_ws in WMMA-native B-fragment order
//    (lane-contiguous 32B per fragment) -> 2x global_load_b128, coalesced,
//    streamed from L2 every step (3.5 MB resident in 192 MB L2).
//  * Base pointers are laundered through an empty asm each timestep so LICM
//    cannot hoist the weight loads out of the T loop; the B-fragment loader
//    casts to address space 1 so the loads stay GLOBAL (not FLAT -> avoids
//    DScnt coupling of the weight stream with LDS fragment loads).
//  * x read and out written with non-temporal hints to keep weights in L2.
// ---------------------------------------------------------------------------

typedef __bf16 bf16_t;
typedef __attribute__((ext_vector_type(16))) __bf16 v16bf;
typedef __attribute__((ext_vector_type(8)))  __bf16 v8bf;
typedef __attribute__((ext_vector_type(8)))  float  v8f;
typedef __attribute__((ext_vector_type(4)))  float  v4f;

// global-address-space vector pointer (forces global_load_b128, not flat)
typedef __attribute__((address_space(1))) const v8bf* gv8bf_p;

#define B_SZ 256
#define T_SZ 512
#define D_SZ 256
#define U_SZ 512
#define DPAD 264   // 256 + 8 pad: keeps 16B alignment, rotates LDS banks
#define HPAD 520   // 512 + 8

__device__ __forceinline__ v16bf cat8(v8bf lo, v8bf hi) {
    return __builtin_shufflevector(lo, hi, 0,1,2,3,4,5,6,7,8,9,10,11,12,13,14,15);
}

// A fragment (16x32 bf16), source row-major [16][stride] in LDS.
// ISA layout: lanes 0-15 (hi=0): slots 0-7 -> K 0-7, slots 8-15 -> K 16-23;
//             lanes 16-31 (hi=1): K 8-15 and 24-31.
__device__ __forceinline__ v16bf load_a_frag(const bf16_t* base, int stride,
                                             int row, int kbase, int hi) {
    const bf16_t* p = base + row * stride + kbase + hi * 8;
    v8bf lo = *reinterpret_cast<const v8bf*>(p);        // ds_load_b128
    v8bf hh = *reinterpret_cast<const v8bf*>(p + 16);   // ds_load_b128
    return cat8(lo, hh);
}

// B fragment (32x16 bf16) pre-packed: lane-contiguous 16 bf16 (32 bytes).
// Explicit AS1 cast keeps these as global_load_b128 even though the base
// pointer was laundered (laundering makes it opaque -> would become flat).
__device__ __forceinline__ v16bf load_b_frag(const bf16_t* blk, int lane) {
    gv8bf_p p = (gv8bf_p)(const void*)(blk + lane * 16);
    v8bf lo = p[0];                                     // global_load_b128
    v8bf hh = p[1];                                     // global_load_b128
    return cat8(lo, hh);
}

__device__ __forceinline__ v8f wmma_bf16(v16bf a, v16bf b, v8f c) {
    return __builtin_amdgcn_wmma_f32_16x16x32_bf16(
        /*neg_a=*/false, a, /*neg_b=*/false, b,
        /*c_mod=*/(short)0, c, /*reuse_a=*/false, /*reuse_b=*/false);
}

__device__ __forceinline__ float sigmoidf_(float x) {
    return 1.0f / (1.0f + __expf(-x));
}

// ---------------------------------------------------------------------------
// Pack one fp32 [K,N] row-major weight into WMMA-native bf16 B-fragment order:
// dst[((nt*numK + kt)*512) + lane*16 + s]  <-  src[k*N + n]
//   k = kt*32 + (lane>>4)*8 + (s&7) + (s>>3)*16,  n = nt*16 + (lane&15)
// Packed size == K*N bf16 elements.
// ---------------------------------------------------------------------------
__global__ void pack_weight_kernel(const float* __restrict__ src,
                                   bf16_t* __restrict__ dst, int K, int N) {
    const int  numK  = K >> 5;
    const long total = (long)K * (long)N;
    for (long e = blockIdx.x * (long)blockDim.x + threadIdx.x; e < total;
         e += (long)gridDim.x * blockDim.x) {
        const int  within = (int)(e & 511);
        const long blk    = e >> 9;
        const int  kt     = (int)(blk % numK);
        const int  nt     = (int)(blk / numK);
        const int  l      = within >> 4;
        const int  s      = within & 15;
        const int  k      = kt * 32 + ((l >> 4) << 3) + (s & 7) + ((s >> 3) << 4);
        const int  n      = nt * 16 + (l & 15);
        dst[e] = (bf16_t)src[(size_t)k * N + n];
    }
}

// ---------------------------------------------------------------------------
// Main recurrent kernel: grid = 16 blocks (batch tiles), block = 512 threads.
// ---------------------------------------------------------------------------
__global__ __launch_bounds__(512)
void alstm_scan_kernel(const float* __restrict__ x,
                       const bf16_t* __restrict__ pwa, const bf16_t* __restrict__ pua,
                       const bf16_t* __restrict__ pwi, const bf16_t* __restrict__ pui,
                       const bf16_t* __restrict__ pwf, const bf16_t* __restrict__ puf,
                       const bf16_t* __restrict__ pwo, const bf16_t* __restrict__ puo,
                       const bf16_t* __restrict__ pwc, const bf16_t* __restrict__ puc,
                       const float* __restrict__ bi, const float* __restrict__ bfv,
                       const float* __restrict__ bo, const float* __restrict__ bc,
                       const float* __restrict__ ba,
                       float* __restrict__ out) {
    extern __shared__ char smem[];
    float*  xt_f32 = (float*)smem;                       // [16][DPAD] fp32 x_t
    float*  attb   = xt_f32 + 16 * DPAD;                 // [16][DPAD] attn preact
    float*  cst    = attb   + 16 * DPAD;                 // [16][U]    cell state
    bf16_t* xt_bf  = (bf16_t*)(cst + 16 * U_SZ);         // [16][DPAD]
    bf16_t* ctx_bf = xt_bf  + 16 * DPAD;                 // [16][DPAD]
    bf16_t* h_bf   = ctx_bf + 16 * DPAD;                 // [2][16][HPAD]

    const int tid   = threadIdx.x;
    const int lane  = tid & 31;
    const int wid   = tid >> 5;          // 0..15
    const int hi    = lane >> 4;         // 0/1
    const int arow  = lane & 15;         // A-matrix row for fragment loads
    const int bbase = blockIdx.x * 16;   // first batch row of this tile

    // init state: c = 0, h = 0 (both LDS buffers)
    for (int i = tid; i < 16 * U_SZ; i += 512) cst[i] = 0.0f;
    for (int i = tid; i < 2 * 16 * HPAD; i += 512) h_bf[i] = (bf16_t)0.0f;
    __syncthreads();

    // Local (launder-able) copies of the packed-weight base pointers.
    const bf16_t* vwa = pwa; const bf16_t* vua = pua;
    const bf16_t* vwi = pwi; const bf16_t* vui = pui;
    const bf16_t* vwf = pwf; const bf16_t* vuf = puf;
    const bf16_t* vwo = pwo; const bf16_t* vuo = puo;
    const bf16_t* vwc = pwc; const bf16_t* vuc = puc;

    int cur = 0;
    for (int t = 0; t < T_SZ; ++t) {
        // Launder weight pointers: zero-cost, defeats LICM so the weight
        // fragment loads stay inside the T loop (stream from L2, no scratch
        // spill of hoisted fragments).
        asm volatile("" : "+s"(vwa), "+s"(vua),
                          "+s"(vwi), "+s"(vui), "+s"(vwf), "+s"(vuf),
                          "+s"(vwo), "+s"(vuo), "+s"(vwc), "+s"(vuc));
        const bf16_t* wg[4] = {vwi, vwf, vwo, vwc};
        const bf16_t* ug[4] = {vui, vuf, vuo, vuc};

        // ------- load x_t tile: one wave per batch row, 8 floats per lane ----
        {
            const int    row = wid;
            const float* src = x + ((size_t)(bbase + row) * T_SZ + t) * D_SZ;
            v4f v0 = __builtin_nontemporal_load(&((const v4f*)src)[lane]);
            v4f v1 = __builtin_nontemporal_load(&((const v4f*)src)[lane + 32]);
            float* drow = xt_f32 + row * DPAD;
            ((v4f*)drow)[lane]         = v0;
            ((v4f*)(drow + 128))[lane] = v1;
            bf16_t* brow = xt_bf + row * DPAD;
            const int d0 = lane * 4;
            brow[d0 + 0] = (bf16_t)v0[0]; brow[d0 + 1] = (bf16_t)v0[1];
            brow[d0 + 2] = (bf16_t)v0[2]; brow[d0 + 3] = (bf16_t)v0[3];
            brow[128 + d0 + 0] = (bf16_t)v1[0]; brow[128 + d0 + 1] = (bf16_t)v1[1];
            brow[128 + d0 + 2] = (bf16_t)v1[2]; brow[128 + d0 + 3] = (bf16_t)v1[3];
            if (t + 1 < T_SZ)
                __builtin_prefetch(src + D_SZ + lane * 8, 0, 3); // next x_t row
        }
        __syncthreads();  // B1: x_t ready; prev-step h writes ordered

        // ------- attention preact: wave wid owns N-tile wid (cols wid*16..) --
        {
            const bf16_t* hcur = h_bf + cur * (16 * HPAD);
            v8f acc = {0.f, 0.f, 0.f, 0.f, 0.f, 0.f, 0.f, 0.f};
            #pragma unroll
            for (int kf = 0; kf < 8; ++kf) {              // x_t @ W_a  (K=256)
                v16bf a = load_a_frag(xt_bf, DPAD, arow, kf * 32, hi);
                v16bf b = load_b_frag(vwa + (((size_t)wid * 8 + kf) << 9), lane);
                acc = wmma_bf16(a, b, acc);
            }
            #pragma unroll
            for (int kf = 0; kf < 16; ++kf) {             // h @ U_a    (K=512)
                v16bf a = load_a_frag(hcur, HPAD, arow, kf * 32, hi);
                v16bf b = load_b_frag(vua + (((size_t)wid * 16 + kf) << 9), lane);
                acc = wmma_bf16(a, b, acc);
            }
            const int   col  = wid * 16 + (lane & 15);
            const float bias = ba[col];
            const int   m0   = hi * 8;
            #pragma unroll
            for (int r = 0; r < 8; ++r)
                attb[(m0 + r) * DPAD + col] = tanhf(acc[r] + bias);
        }
        __syncthreads();  // B2: attention preacts ready

        // ------- softmax over D per row (wave wid owns row wid), build ctx ---
        {
            const int row = wid;
            float v[8];
            float mx = -3.0e38f;
            #pragma unroll
            for (int j = 0; j < 8; ++j) {
                v[j] = attb[row * DPAD + lane + 32 * j];
                mx = fmaxf(mx, v[j]);
            }
            for (int off = 16; off; off >>= 1)
                mx = fmaxf(mx, __shfl_xor(mx, off, 32));
            float s = 0.f;
            #pragma unroll
            for (int j = 0; j < 8; ++j) { v[j] = __expf(v[j] - mx); s += v[j]; }
            for (int off = 16; off; off >>= 1) s += __shfl_xor(s, off, 32);
            const float inv = 1.0f / s;
            #pragma unroll
            for (int j = 0; j < 8; ++j) {
                const int d = lane + 32 * j;
                ctx_bf[row * DPAD + d] =
                    (bf16_t)(xt_f32[row * DPAD + d] * v[j] * inv);
            }
        }
        __syncthreads();  // B3: ctx ready

        // ------- gates: wave wid owns N-tiles {2*wid, 2*wid+1} x 4 gates -----
        {
            const bf16_t* hcur = h_bf + cur * (16 * HPAD);
            bf16_t*       hnxt = h_bf + (cur ^ 1) * (16 * HPAD);
            const int     nt0  = wid * 2;

            v8f acc[4][2];
            #pragma unroll
            for (int g = 0; g < 4; ++g)
                #pragma unroll
                for (int tt = 0; tt < 2; ++tt)
                    acc[g][tt] = (v8f){0.f, 0.f, 0.f, 0.f, 0.f, 0.f, 0.f, 0.f};

            // K = 256 (ctx) + 512 (h) = 24 fragments; A reused across 8 tiles.
            for (int kf = 0; kf < 24; ++kf) {
                v16bf a = (kf < 8)
                    ? load_a_frag(ctx_bf, DPAD, arow, kf * 32, hi)
                    : load_a_frag(hcur,  HPAD, arow, (kf - 8) * 32, hi);
                #pragma unroll
                for (int g = 0; g < 4; ++g) {
                    #pragma unroll
                    for (int tt = 0; tt < 2; ++tt) {
                        const bf16_t* bp = (kf < 8)
                            ? wg[g] + (((size_t)(nt0 + tt) * 8 + kf) << 9)
                            : ug[g] + (((size_t)(nt0 + tt) * 16 + (kf - 8)) << 9);
                        acc[g][tt] = wmma_bf16(a, load_b_frag(bp, lane), acc[g][tt]);
                    }
                }
            }

            // elementwise LSTM update in-register (C/D: m = r + 8*hi, n = lane&15)
            #pragma unroll
            for (int tt = 0; tt < 2; ++tt) {
                const int   u   = (nt0 + tt) * 16 + (lane & 15);
                const float vbi = bi[u], vbf = bfv[u], vbo = bo[u], vbc = bc[u];
                #pragma unroll
                for (int r = 0; r < 8; ++r) {
                    const int   m  = hi * 8 + r;
                    const float iv = sigmoidf_(acc[0][tt][r] + vbi);
                    const float fv = sigmoidf_(acc[1][tt][r] + vbf);
                    const float ov = sigmoidf_(acc[2][tt][r] + vbo);
                    const float cv = tanhf(acc[3][tt][r] + vbc);
                    const float cn = fv * cst[m * U_SZ + u] + iv * cv;
                    cst[m * U_SZ + u] = cn;
                    const float hn = ov * tanhf(cn);
                    hnxt[m * HPAD + u] = (bf16_t)hn;
                    __builtin_nontemporal_store(
                        hn, &out[((size_t)(bbase + m) * T_SZ + t) * U_SZ + u]);
                }
            }
        }
        cur ^= 1;
        // next-iteration B1 orders hnxt writes vs. phase-A reads
    }
}

// ---------------------------------------------------------------------------
extern "C" void kernel_launch(void* const* d_in, const int* in_sizes, int n_in,
                              void* d_out, int out_size, void* d_ws, size_t ws_size,
                              hipStream_t stream) {
    const float* x   = (const float*)d_in[0];
    const float* W_i = (const float*)d_in[1];
    const float* U_i = (const float*)d_in[2];
    const float* b_i = (const float*)d_in[3];
    const float* W_f = (const float*)d_in[4];
    const float* U_f = (const float*)d_in[5];
    const float* b_f = (const float*)d_in[6];
    const float* W_o = (const float*)d_in[7];
    const float* U_o = (const float*)d_in[8];
    const float* b_o = (const float*)d_in[9];
    const float* W_c = (const float*)d_in[10];
    const float* U_c = (const float*)d_in[11];
    const float* b_c = (const float*)d_in[12];
    const float* W_a = (const float*)d_in[13];
    const float* U_a = (const float*)d_in[14];
    const float* b_a = (const float*)d_in[15];
    float* out = (float*)d_out;

    // workspace: packed bf16 weights (packed size == K*N elements each)
    bf16_t* ws  = (bf16_t*)d_ws;
    size_t  off = 0;
    bf16_t* pWa = ws + off; off += (size_t)D_SZ * D_SZ;   // 256x256
    bf16_t* pUa = ws + off; off += (size_t)U_SZ * D_SZ;   // 512x256
    bf16_t* pWi = ws + off; off += (size_t)D_SZ * U_SZ;
    bf16_t* pWf = ws + off; off += (size_t)D_SZ * U_SZ;
    bf16_t* pWo = ws + off; off += (size_t)D_SZ * U_SZ;
    bf16_t* pWc = ws + off; off += (size_t)D_SZ * U_SZ;
    bf16_t* pUi = ws + off; off += (size_t)U_SZ * U_SZ;
    bf16_t* pUf = ws + off; off += (size_t)U_SZ * U_SZ;
    bf16_t* pUo = ws + off; off += (size_t)U_SZ * U_SZ;
    bf16_t* pUc = ws + off; off += (size_t)U_SZ * U_SZ;

    struct { const float* s; bf16_t* d; int K, N; } jobs[10] = {
        {W_a, pWa, D_SZ, D_SZ}, {U_a, pUa, U_SZ, D_SZ},
        {W_i, pWi, D_SZ, U_SZ}, {W_f, pWf, D_SZ, U_SZ},
        {W_o, pWo, D_SZ, U_SZ}, {W_c, pWc, D_SZ, U_SZ},
        {U_i, pUi, U_SZ, U_SZ}, {U_f, pUf, U_SZ, U_SZ},
        {U_o, pUo, U_SZ, U_SZ}, {U_c, pUc, U_SZ, U_SZ},
    };
    for (int j = 0; j < 10; ++j) {
        long total = (long)jobs[j].K * jobs[j].N;
        int  grid  = (int)((total + 255) / 256);
        pack_weight_kernel<<<grid, 256, 0, stream>>>(jobs[j].s, jobs[j].d,
                                                     jobs[j].K, jobs[j].N);
    }

    const size_t smem_bytes =
        (size_t)16 * DPAD * 4 +      // xt fp32
        (size_t)16 * DPAD * 4 +      // attn preact fp32
        (size_t)16 * U_SZ * 4 +      // c fp32
        (size_t)16 * DPAD * 2 +      // xt bf16
        (size_t)16 * DPAD * 2 +      // ctx bf16
        (size_t)2 * 16 * HPAD * 2;   // h bf16 double buffer  (~114 KB total)

    alstm_scan_kernel<<<B_SZ / 16, 512, smem_bytes, stream>>>(
        x, pWa, pUa, pWi, pUi, pWf, pUf, pWo, pUo, pWc, pUc,
        b_i, b_f, b_o, b_c, b_a, out);
}
